// SiTBlock_57019985822465
// MI455X (gfx1250) — matrix operand order repaired and checked by reading, not compiled, not run on hardware
//
#include <hip/hip_runtime.h>
#include <cstdint>
#include <cstddef>

typedef unsigned short u16;
typedef __attribute__((ext_vector_type(16))) __bf16 v16bf;
typedef __attribute__((ext_vector_type(8)))  float  v8f;

union Frag { v16bf v; uint4 q[2]; };
union Acc8 { v8f v; float f[8]; };

static __device__ __forceinline__ u16 f2bf(float f) {
    unsigned u = __float_as_uint(f);
    u += 0x7FFFu + ((u >> 16) & 1u);   // round-to-nearest-even
    return (u16)(u >> 16);
}

// Async global->LDS copy, 16B per lane (CDNA5 GLOBAL_LOAD_ASYNC_TO_LDS_B128,
// tracked by ASYNCcnt). lds_addr = low 32 bits of the generic pointer (LDS
// aperture offset per ISA section 10.2).
static __device__ __forceinline__ void async_cp_b128(void* lds_ptr, const void* gptr) {
    unsigned lds_addr = (unsigned)(uintptr_t)lds_ptr;
    asm volatile("global_load_async_to_lds_b128 %0, %1, off"
                 :: "v"(lds_addr), "v"(gptr)
                 : "memory");
}
static __device__ __forceinline__ void wait_async0() {
    asm volatile("s_wait_asynccnt 0x0" ::: "memory");
}

// ---------------------------------------------------------------------------
// small kernels
// ---------------------------------------------------------------------------
__global__ void silu_kernel(const float* __restrict__ c, float* __restrict__ out, int n) {
    int i = blockIdx.x * 256 + threadIdx.x;
    if (i < n) { float x = c[i]; out[i] = x / (1.f + __expf(-x)); }
}

// mod[b,6C] = silu(c) @ ada_w^T + ada_b   (8 x 6144, K=1024)
__global__ void ada_kernel(const float* __restrict__ sc, const float* __restrict__ w,
                           const float* __restrict__ b, float* __restrict__ mod) {
    int g  = blockIdx.x * 256 + threadIdx.x;           // [0, 8*6144)
    int bb = g / 6144, j = g % 6144;
    const float4* wp = (const float4*)(w + (size_t)j * 1024);
    const float4* sp = (const float4*)(sc + bb * 1024);
    float acc = 0.f;
    #pragma unroll 4
    for (int k = 0; k < 256; ++k) {
        float4 a = wp[k], x = sp[k];
        acc += a.x * x.x + a.y * x.y + a.z * x.z + a.w * x.w;
    }
    mod[g] = acc + b[j];
}

__global__ void cvt_bf16_kernel(const float* __restrict__ src, u16* __restrict__ dst, int n) {
    int i = (blockIdx.x * 256 + threadIdx.x) * 4;
    if (i < n) {
        float4 v = *(const float4*)(src + i);
        unsigned lo = (unsigned)f2bf(v.x) | ((unsigned)f2bf(v.y) << 16);
        unsigned hi = (unsigned)f2bf(v.z) | ((unsigned)f2bf(v.w) << 16);
        uint2 t; t.x = lo; t.y = hi;
        *(uint2*)(dst + i) = t;
    }
}

// layernorm(x[r,:]) * (1+scale[b,:]) + shift[b,:]  -> bf16 h   (one block per row)
__global__ void ln_mod_kernel(const float* __restrict__ x, const float* __restrict__ mod,
                              int shOff, int scOff, u16* __restrict__ h) {
    const int r = blockIdx.x, tid = threadIdx.x;
    const int b = r >> 10;
    const float* xr = x + (size_t)r * 1024;
    float4 xv = *(const float4*)(xr + tid * 4);
    float s  = xv.x + xv.y + xv.z + xv.w;
    float sq = xv.x * xv.x + xv.y * xv.y + xv.z * xv.z + xv.w * xv.w;
    #pragma unroll
    for (int m = 1; m < 32; m <<= 1) { s += __shfl_xor(s, m, 32); sq += __shfl_xor(sq, m, 32); }
    __shared__ float a1[8], a2[8];
    int lane = tid & 31, wid = tid >> 5;
    if (lane == 0) { a1[wid] = s; a2[wid] = sq; }
    __syncthreads();
    float ts = 0.f, tq = 0.f;
    #pragma unroll
    for (int i = 0; i < 8; ++i) { ts += a1[i]; tq += a2[i]; }
    float mu = ts * (1.f / 1024.f);
    float var = tq * (1.f / 1024.f) - mu * mu;
    float rs = rsqrtf(var + 1e-6f);
    int cb = tid * 4;
    u16 o[4];
    #pragma unroll
    for (int e = 0; e < 4; ++e) {
        int cc = cb + e;
        float xe = (e == 0) ? xv.x : (e == 1) ? xv.y : (e == 2) ? xv.z : xv.w;
        float scl = mod[b * 6144 + scOff + cc];
        float sh  = mod[b * 6144 + shOff + cc];
        o[e] = f2bf((xe - mu) * rs * (1.f + scl) + sh);
    }
    unsigned lo = (unsigned)o[0] | ((unsigned)o[1] << 16);
    unsigned hi = (unsigned)o[2] | ((unsigned)o[3] << 16);
    uint2 t; t.x = lo; t.y = hi;
    *(uint2*)(h + (size_t)r * 1024 + cb) = t;
}

// ---------------------------------------------------------------------------
// WMMA GEMM: C[M,N] = A[M,K](bf16 rowmajor) * Bw[N,K](bf16 rowmajor, i.e. W) + bias
// BM=BN=128, BK=32; 256 threads = 8 waves, each wave 64x32 (4x2 tiles of 16x16).
// Tiles staged with GLOBAL_LOAD_ASYNC_TO_LDS_B128 (double buffered, ASYNCcnt).
// EPI: 0=qkv split, 1=proj+residual+gate, 2=gelu->bf16, 3=fc2 gate accumulate
// ---------------------------------------------------------------------------
template <int EPI>
__launch_bounds__(256)
__global__ void gemm_bf16_kernel(const u16* __restrict__ A, const u16* __restrict__ Bw,
                                 const float* __restrict__ bias, int M, int N, int K,
                                 u16* __restrict__ outQ, u16* __restrict__ outK, u16* __restrict__ outV,
                                 float* __restrict__ outF, const float* __restrict__ xin,
                                 const float* __restrict__ mod, u16* __restrict__ outH) {
    __shared__ __align__(16) u16 As[2][128 * 40];
    __shared__ __align__(16) u16 Bs[2][128 * 40];
    const int tid = threadIdx.x, lane = tid & 31, wid = tid >> 5;
    const int wr = wid >> 2, wc = wid & 3;
    const int m0 = blockIdx.y * 128, n0 = blockIdx.x * 128;
    const int KT = K >> 5;

    Acc8 acc[4][2];
    #pragma unroll
    for (int i = 0; i < 4; ++i)
        #pragma unroll
        for (int j = 0; j < 2; ++j)
            #pragma unroll
            for (int v = 0; v < 8; ++v) acc[i][j].f[v] = 0.f;

    // per-thread copy geometry: 2 chunks of 16B for A tile, 2 for B tile
    const int row0 = tid >> 2,           ch0 = tid & 3;
    const int row1 = (256 + tid) >> 2,   ch1 = (256 + tid) & 3;

    auto issue_tile = [&](int kt, int buf) {
        int kb = kt * 32;
        async_cp_b128(&As[buf][row0 * 40 + ch0 * 8], A  + (size_t)(m0 + row0) * K + kb + ch0 * 8);
        async_cp_b128(&As[buf][row1 * 40 + ch1 * 8], A  + (size_t)(m0 + row1) * K + kb + ch1 * 8);
        async_cp_b128(&Bs[buf][row0 * 40 + ch0 * 8], Bw + (size_t)(n0 + row0) * K + kb + ch0 * 8);
        async_cp_b128(&Bs[buf][row1 * 40 + ch1 * 8], Bw + (size_t)(n0 + row1) * K + kb + ch1 * 8);
    };

    issue_tile(0, 0);

    for (int kt = 0; kt < KT; ++kt) {
        int cur = kt & 1;
        wait_async0();        // own async copies into buf `cur` done
        __syncthreads();      // everyone's done; previous readers of cur^1 done
        if (kt + 1 < KT) issue_tile(kt + 1, cur ^ 1);

        Frag bfr[2];
        #pragma unroll
        for (int j = 0; j < 2; ++j) {
            const u16* bp = &Bs[cur][(wc * 32 + j * 16 + (lane & 15)) * 40 + (lane >> 4) * 8];
            bfr[j].q[0] = *(const uint4*)(bp);
            bfr[j].q[1] = *(const uint4*)(bp + 16);
        }
        Frag afr[4];
        #pragma unroll
        for (int i = 0; i < 4; ++i) {
            const u16* ap = &As[cur][(wr * 64 + i * 16 + (lane & 15)) * 40 + (lane >> 4) * 8];
            afr[i].q[0] = *(const uint4*)(ap);
            afr[i].q[1] = *(const uint4*)(ap + 16);
        }
        #pragma unroll
        for (int i = 0; i < 4; ++i)
            #pragma unroll
            for (int j = 0; j < 2; ++j)
                acc[i][j].v = __builtin_amdgcn_wmma_f32_16x16x32_bf16(
                    false, afr[i].v, false, bfr[j].v, (short)0, acc[i][j].v, false, false);
    }

    // epilogue
    const int half = lane >> 4;
    #pragma unroll
    for (int i = 0; i < 4; ++i)
        #pragma unroll
        for (int j = 0; j < 2; ++j)
            #pragma unroll
            for (int v = 0; v < 8; ++v) {
                int m = m0 + wr * 64 + i * 16 + v + 8 * half;
                int n = n0 + wc * 32 + j * 16 + (lane & 15);
                float val = acc[i][j].f[v] + bias[n];
                if constexpr (EPI == 0) {
                    int which = n >> 10, nh = n & 1023;
                    int hh = nh >> 6, d = nh & 63;
                    int bb = m >> 10, sm = m & 1023;
                    size_t idx = (((size_t)(bb * 16 + hh)) * 1024 + sm) * 64 + d;
                    if (which == 0)      outQ[idx] = f2bf(val * 0.125f);  // fold HD^-0.5
                    else if (which == 1) outK[idx] = f2bf(val);
                    else                 outV[idx] = f2bf(val);
                } else if constexpr (EPI == 1) {
                    int bb = m >> 10;
                    float g = mod[bb * 6144 + 2048 + n];                  // gate_msa
                    size_t idx = (size_t)m * 1024 + n;
                    outF[idx] = xin[idx] + g * val;
                } else if constexpr (EPI == 2) {
                    float t = 0.7978845608028654f * (val + 0.044715f * val * val * val);
                    float gl = 0.5f * val * (1.f + tanhf(t));
                    outH[(size_t)m * 4096 + n] = f2bf(gl);
                } else {
                    int bb = m >> 10;
                    float g = mod[bb * 6144 + 5120 + n];                  // gate_mlp
                    size_t idx = (size_t)m * 1024 + n;
                    outF[idx] = outF[idx] + g * val;
                }
            }
}

// ---------------------------------------------------------------------------
// Flash attention: block = 128 thr (4 waves), handles (b,h, 64-row q block).
// Wave w owns 16 q rows. S=Q*K^T (WMMA), online softmax, O += P*V (WMMA).
// Q/K tiles staged with async-to-LDS; V staged transposed via ds stores.
// ---------------------------------------------------------------------------
__launch_bounds__(128)
__global__ void attn_kernel(const u16* __restrict__ qb_, const u16* __restrict__ kb_,
                            const u16* __restrict__ vb_, const unsigned char* __restrict__ pm,
                            u16* __restrict__ attn) {
    __shared__ __align__(16) u16 Qs[64 * 72];
    __shared__ __align__(16) u16 Ks[128 * 72];
    __shared__ __align__(16) u16 Vt[64 * 136];
    __shared__ __align__(16) u16 Ps[64 * 136];
    const int tid = threadIdx.x, lane = tid & 31, w = tid >> 5;
    const int bh = blockIdx.x >> 4, qb = blockIdx.x & 15;
    const int b = bh >> 4, hh = bh & 15;
    const int qm0 = qb * 64;
    const int half = lane >> 4;
    const size_t base = (size_t)bh * 1024 * 64;

    // stage Q (64 x 64 bf16) asynchronously
    const u16* qp = qb_ + base + (size_t)qm0 * 64;
    #pragma unroll
    for (int p = 0; p < 4; ++p) {
        int lin = p * 128 + tid;
        int row = lin >> 3, ch = lin & 7;
        async_cp_b128(&Qs[row * 72 + ch * 8], qp + row * 64 + ch * 8);
    }

    float rm[8], rl[8];
    bool  rv[8];
    #pragma unroll
    for (int v = 0; v < 8; ++v) {
        rm[v] = -3.0e38f; rl[v] = 0.f;
        rv[v] = pm[b * 1024 + qm0 + w * 16 + v + 8 * half] != 0;
    }
    Acc8 o[4];
    #pragma unroll
    for (int jo = 0; jo < 4; ++jo)
        #pragma unroll
        for (int v = 0; v < 8; ++v) o[jo].f[v] = 0.f;

    wait_async0();
    __syncthreads();

    // preload Q frags (wave-constant)
    Frag qf[2];
    #pragma unroll
    for (int kk = 0; kk < 2; ++kk) {
        const u16* ap = &Qs[(w * 16 + (lane & 15)) * 72 + kk * 32 + half * 8];
        qf[kk].q[0] = *(const uint4*)(ap);
        qf[kk].q[1] = *(const uint4*)(ap + 16);
    }

    for (int jb = 0; jb < 8; ++jb) {
        __syncthreads();   // previous iteration's readers of Ks/Vt are done
        const u16* kp = kb_ + base + (size_t)jb * 128 * 64;
        #pragma unroll
        for (int p = 0; p < 8; ++p) {
            int lin = p * 128 + tid;
            int row = lin >> 3, ch = lin & 7;
            async_cp_b128(&Ks[row * 72 + ch * 8], kp + row * 64 + ch * 8);
        }
        const u16* vp = vb_ + base + (size_t)jb * 128 * 64;
        #pragma unroll
        for (int p = 0; p < 8; ++p) {
            int lin = p * 128 + tid;
            int row = lin >> 3, ch = lin & 7;
            union { uint4 q; u16 s[8]; } u;
            u.q = *(const uint4*)(vp + row * 64 + ch * 8);
            #pragma unroll
            for (int e = 0; e < 8; ++e) Vt[(ch * 8 + e) * 136 + row] = u.s[e];
        }
        wait_async0();
        __syncthreads();

        // S = Q * K^T for this 64x128 stripe (wave's 16 rows x 128 cols)
        Acc8 s[8];
        #pragma unroll
        for (int jj = 0; jj < 8; ++jj) {
            #pragma unroll
            for (int v = 0; v < 8; ++v) s[jj].f[v] = 0.f;
            #pragma unroll
            for (int kk = 0; kk < 2; ++kk) {
                Frag kf;
                const u16* bp = &Ks[(jj * 16 + (lane & 15)) * 72 + kk * 32 + half * 8];
                kf.q[0] = *(const uint4*)(bp);
                kf.q[1] = *(const uint4*)(bp + 16);
                s[jj].v = __builtin_amdgcn_wmma_f32_16x16x32_bf16(
                    false, qf[kk].v, false, kf.v, (short)0, s[jj].v, false, false);
            }
            bool cv = pm[b * 1024 + jb * 128 + jj * 16 + (lane & 15)] != 0;
            #pragma unroll
            for (int v = 0; v < 8; ++v)
                if (!(cv && rv[v])) s[jj].f[v] += -1.0e9f;
        }

        // online softmax per row, P -> LDS (bf16, A-operand layout via round trip)
        #pragma unroll
        for (int v = 0; v < 8; ++v) {
            float tm = s[0].f[v];
            #pragma unroll
            for (int jj = 1; jj < 8; ++jj) tm = fmaxf(tm, s[jj].f[v]);
            tm = fmaxf(tm, __shfl_xor(tm, 1, 32));
            tm = fmaxf(tm, __shfl_xor(tm, 2, 32));
            tm = fmaxf(tm, __shfl_xor(tm, 4, 32));
            tm = fmaxf(tm, __shfl_xor(tm, 8, 32));
            float nm = fmaxf(rm[v], tm);
            float alpha = __expf(rm[v] - nm);
            rm[v] = nm;
            float ps = 0.f;
            #pragma unroll
            for (int jj = 0; jj < 8; ++jj) {
                float p = __expf(s[jj].f[v] - nm);
                s[jj].f[v] = p;
                ps += p;
            }
            ps += __shfl_xor(ps, 1, 32);
            ps += __shfl_xor(ps, 2, 32);
            ps += __shfl_xor(ps, 4, 32);
            ps += __shfl_xor(ps, 8, 32);
            rl[v] = rl[v] * alpha + ps;
            #pragma unroll
            for (int jo = 0; jo < 4; ++jo) o[jo].f[v] *= alpha;
            int prow = w * 16 + v + 8 * half;
            #pragma unroll
            for (int jj = 0; jj < 8; ++jj)
                Ps[prow * 136 + jj * 16 + (lane & 15)] = f2bf(s[jj].f[v]);
        }

        // O += P * V (wave-private P rows; DS ops are in-order per wave)
        #pragma unroll
        for (int kk = 0; kk < 4; ++kk) {
            Frag pf;
            const u16* ap = &Ps[(w * 16 + (lane & 15)) * 136 + kk * 32 + half * 8];
            pf.q[0] = *(const uint4*)(ap);
            pf.q[1] = *(const uint4*)(ap + 16);
            #pragma unroll
            for (int jo = 0; jo < 4; ++jo) {
                Frag vf;
                const u16* bp = &Vt[(jo * 16 + (lane & 15)) * 136 + kk * 32 + half * 8];
                vf.q[0] = *(const uint4*)(bp);
                vf.q[1] = *(const uint4*)(bp + 16);
                o[jo].v = __builtin_amdgcn_wmma_f32_16x16x32_bf16(
                    false, pf.v, false, vf.v, (short)0, o[jo].v, false, false);
            }
        }
    }

    // write O / l  ->  attn[b*N + r, hh*64 + d]  (bf16)
    #pragma unroll
    for (int jo = 0; jo < 4; ++jo)
        #pragma unroll
        for (int v = 0; v < 8; ++v) {
            int r = qm0 + w * 16 + v + 8 * half;
            float val = o[jo].f[v] / rl[v];
            attn[((size_t)(b * 1024) + r) * 1024 + hh * 64 + jo * 16 + (lane & 15)] = f2bf(val);
        }
}

// ---------------------------------------------------------------------------
extern "C" void kernel_launch(void* const* d_in, const int* in_sizes, int n_in,
                              void* d_out, int out_size, void* d_ws, size_t ws_size,
                              hipStream_t stream) {
    (void)in_sizes; (void)n_in; (void)out_size; (void)ws_size;
    const float* x      = (const float*)d_in[0];
    const float* c      = (const float*)d_in[1];
    const unsigned char* pm = (const unsigned char*)d_in[2];
    const float* qkv_w  = (const float*)d_in[3];
    const float* qkv_b  = (const float*)d_in[4];
    const float* proj_w = (const float*)d_in[5];
    const float* proj_b = (const float*)d_in[6];
    const float* fc1_w  = (const float*)d_in[7];
    const float* fc1_b  = (const float*)d_in[8];
    const float* fc2_w  = (const float*)d_in[9];
    const float* fc2_b  = (const float*)d_in[10];
    const float* ada_w  = (const float*)d_in[11];
    const float* ada_b  = (const float*)d_in[12];
    float* out = (float*)d_out;

    char* ws = (char*)d_ws;
    size_t off = 0;
    auto alloc = [&](size_t bytes) -> char* {
        char* p = ws + off;
        off = (off + bytes + 255) & ~(size_t)255;
        return p;
    };
    float* silu_c = (float*)alloc((size_t)8 * 1024 * 4);
    float* mod    = (float*)alloc((size_t)8 * 6144 * 4);
    u16*   h      = (u16*)alloc((size_t)8192 * 1024 * 2);
    u16*   wqkv   = (u16*)alloc((size_t)3072 * 1024 * 2);
    u16*   wproj  = (u16*)alloc((size_t)1024 * 1024 * 2);
    u16*   wfc1   = (u16*)alloc((size_t)4096 * 1024 * 2);
    u16*   wfc2   = (u16*)alloc((size_t)1024 * 4096 * 2);
    u16*   qbuf   = (u16*)alloc((size_t)8192 * 1024 * 2 * 4); // q,k,v,attn (64MB)
    u16*   kbuf   = qbuf + (size_t)8192 * 1024;
    u16*   vbuf   = kbuf + (size_t)8192 * 1024;
    u16*   attn   = vbuf + (size_t)8192 * 1024;
    u16*   hidden = qbuf;  // alias: q/k/v/attn dead once proj has run

    // 1. adaLN modulation
    silu_kernel<<<32, 256, 0, stream>>>(c, silu_c, 8192);
    ada_kernel<<<192, 256, 0, stream>>>(silu_c, ada_w, ada_b, mod);

    // 2. weights -> bf16 ([N,K] row-major == WMMA B-operand layout)
    cvt_bf16_kernel<<<3072, 256, 0, stream>>>(qkv_w, wqkv, 3072 * 1024);
    cvt_bf16_kernel<<<1024, 256, 0, stream>>>(proj_w, wproj, 1024 * 1024);
    cvt_bf16_kernel<<<4096, 256, 0, stream>>>(fc1_w, wfc1, 4096 * 1024);
    cvt_bf16_kernel<<<4096, 256, 0, stream>>>(fc2_w, wfc2, 4096 * 1024);

    // 3. LN + modulate (msa) -> h
    ln_mod_kernel<<<8192, 256, 0, stream>>>(x, mod, 0, 1024, h);

    // 4. qkv = h @ qkv_w^T + b  (8192x3072, K=1024), scatter to [B,H,N,HD]
    gemm_bf16_kernel<0><<<dim3(24, 64), 256, 0, stream>>>(
        h, wqkv, qkv_b, 8192, 3072, 1024, qbuf, kbuf, vbuf, nullptr, nullptr, nullptr, nullptr);

    // 5. attention
    attn_kernel<<<2048, 128, 0, stream>>>(qbuf, kbuf, vbuf, pm, attn);

    // 6. proj + gate_msa residual -> d_out (f32)
    gemm_bf16_kernel<1><<<dim3(8, 64), 256, 0, stream>>>(
        attn, wproj, proj_b, 8192, 1024, 1024, nullptr, nullptr, nullptr, out, x, mod, nullptr);

    // 7. LN + modulate (mlp) -> h
    ln_mod_kernel<<<8192, 256, 0, stream>>>(out, mod, 3072, 4096, h);

    // 8. fc1 + GELU -> hidden (bf16)
    gemm_bf16_kernel<2><<<dim3(32, 64), 256, 0, stream>>>(
        h, wfc1, fc1_b, 8192, 4096, 1024, nullptr, nullptr, nullptr, nullptr, nullptr, nullptr, hidden);

    // 9. fc2 + gate_mlp residual accumulate into d_out
    gemm_bf16_kernel<3><<<dim3(8, 64), 256, 0, stream>>>(
        hidden, wfc2, fc2_b, 8192, 1024, 4096, nullptr, nullptr, nullptr, out, nullptr, mod, nullptr);
}
